// GAKFGenerator_38792144618230
// MI455X (gfx1250) — matrix operand order repaired and checked by reading, not compiled, not run on hardware
//
#include <hip/hip_runtime.h>
#include <hip/hip_bf16.h>

#define DEVI __device__ __forceinline__

typedef __bf16 bf16_t;
typedef __attribute__((ext_vector_type(16))) __bf16 v16bf;
typedef __attribute__((ext_vector_type(8)))  __bf16 v8bf;
typedef __attribute__((ext_vector_type(8)))  float  v8f;

constexpr int B_  = 32;
constexpr int T_  = 1024;
constexpr int M_  = 8;
constexpr int N_  = 16;
constexpr int HD_ = 256;
constexpr int G3  = 768;          // 3*HD
constexpr int BT  = B_ * T_;      // 32768
constexpr float COV_EPS_   = 1e-5f;
constexpr float POSDEF_EPS_ = 1e-6f;

// Dynamic LDS for the persistent GRU kernel:
//   h   bf16 [32][256]  = 16384 B
//   gh  f32  [32][768]  = 98304 B
//   xw  f32  [32][768]  = 98304 B   (async-staged per timestep)
constexpr unsigned GRU_H_BYTES  = B_ * HD_ * sizeof(bf16_t);
constexpr unsigned GRU_GH_BYTES = B_ * G3 * sizeof(float);
constexpr unsigned GRU_XW_BYTES = B_ * G3 * sizeof(float);
constexpr unsigned GRU_SMEM = GRU_H_BYTES + GRU_GH_BYTES + GRU_XW_BYTES; // 212992 B

// ---------------------------------------------------------------------------
// WMMA helpers (CDNA5 16x16x32 bf16, f32 accumulate)
// ---------------------------------------------------------------------------
DEVI v8f wmma_bf16(v16bf a, v16bf b, v8f c) {
  return __builtin_amdgcn_wmma_f32_16x16x32_bf16(
      /*neg_a=*/false, a, /*neg_b=*/false, b,
      /*c_mod=*/(short)0, c, /*reuse_a=*/false, /*reuse_b=*/false);
}

// A fragment (16x32 bf16): lane L -> row = base + L%16;
// lanes 0-15 hold K = k0+{0..15}; lanes 16-31 hold K = k0+{16..31}
DEVI v16bf make_a_frag(const bf16_t* row, int k0, int lane) {
  const int kh = (lane >> 4) * 16;
  v8bf lo = *(const v8bf*)(row + k0 + kh);
  v8bf hi = *(const v8bf*)(row + k0 + kh + 8);
  v16bf r;
#pragma unroll
  for (int i = 0; i < 8; ++i) { r[i] = lo[i]; r[8 + i] = hi[i]; }
  return r;
}

DEVI float sigmoidf_(float x) { return 1.f / (1.f + __expf(-x)); }
DEVI float tanhf_(float x) {   // 1 - 2/(e^{2x}+1); saturates to +/-1
  const float e2 = __expf(2.f * x);
  return 1.f - 2.f / (e2 + 1.f);
}

// ---------------------------------------------------------------------------
// Weight prep
// ---------------------------------------------------------------------------
__global__ __launch_bounds__(256) void k_cvt_bf16(bf16_t* __restrict__ dst,
                                                  const float* __restrict__ src, int n) {
  int i = blockIdx.x * 256 + threadIdx.x;
  if (i < n) dst[i] = (bf16_t)src[i];
}

__global__ __launch_bounds__(256) void k_pack_heads(
    bf16_t* __restrict__ whead, float* __restrict__ bhead,
    const float* __restrict__ Wm, const float* __restrict__ Wlv,
    const float* __restrict__ Wres, const float* __restrict__ bm,
    const float* __restrict__ blv, const float* __restrict__ bres) {
  int i = blockIdx.x * 256 + threadIdx.x;
  if (i < 48 * HD_) {
    int r = i >> 8, c = i & 255;
    float v = (r < 16) ? Wm[r * HD_ + c]
            : (r < 32) ? Wlv[(r - 16) * HD_ + c]
                       : Wres[(r - 32) * HD_ + c];
    whead[i] = (bf16_t)v;
  }
  if (i < 48) {
    bhead[i] = (i < 16) ? bm[i] : (i < 32) ? blv[i - 16] : bres[i - 32];
  }
}

// ---------------------------------------------------------------------------
// Layer-0 input projection: xw[bt,g] = y_input[bt,:8] . Wih0[g,:8] + bih0[g]
// y_input[b,t] = (t==0) ? 0 : y_seq[b,t-1]. K=8 -> scalar FMA kernel.
// ---------------------------------------------------------------------------
__global__ __launch_bounds__(256) void k_xw0(
    float* __restrict__ xw, const float* __restrict__ y_seq,
    const float* __restrict__ Wih0, const float* __restrict__ bih0) {
  const int i  = blockIdx.x * 256 + threadIdx.x;   // < BT*G3 = 25,165,824
  const int g  = i % G3;
  const int bt = i / G3;
  const int t  = bt & (T_ - 1);
  float acc = bih0[g];
  if (t > 0) {
    const float* yr = y_seq + (size_t)(bt - 1) * M_;
    const float* wr = Wih0 + (size_t)g * M_;
#pragma unroll
    for (int k = 0; k < M_; ++k) acc += yr[k] * wr[k];
  }
  xw[(size_t)bt * G3 + g] = acc;
}

// ---------------------------------------------------------------------------
// Persistent single-workgroup GRU scan (one layer).
// 1024 threads = 32 waves. Per step:
//   (a) issue async DMA of this step's xw slab (96 KB) into LDS (overlaps GEMM)
//   (b) gh = h @ Whh^T via 96 WMMA tiles (2 M-tiles x 48 N-tiles, K=256)
//   (c) s_wait_asynccnt 0 + barrier, then gate math reading xw from LDS
// h state in LDS (bf16), gh staged in LDS (f32). Whh stays WGP$/L2-hot.
// ---------------------------------------------------------------------------
__global__ __launch_bounds__(1024) void k_gru(
    bf16_t* __restrict__ h_out, const float* __restrict__ xw,
    const bf16_t* __restrict__ Whh, const float* __restrict__ bhh) {
  extern __shared__ char smem[];
  bf16_t* h_s  = (bf16_t*)smem;                          // [32][256]
  float*  gh_s = (float*)(smem + GRU_H_BYTES);           // [32][768]
  float*  xw_s = (float*)(smem + GRU_H_BYTES + GRU_GH_BYTES); // [32][768]

  // raw LDS byte offset of xw_s (dynamic LDS starts after static LDS)
  const unsigned xw_lds_base =
      __builtin_amdgcn_groupstaticsize() + GRU_H_BYTES + GRU_GH_BYTES;

  const int tid  = threadIdx.x;
  const int lane = tid & 31;
  const int wave = tid >> 5;

  for (int i = tid; i < B_ * HD_; i += 1024) h_s[i] = (bf16_t)0.f;
  __syncthreads();

  for (int t = 0; t < T_; ++t) {
    // ---- (a) async DMA xw[t] -> LDS (6144 x b128, 6 per thread) ----
#pragma unroll
    for (int i = 0; i < 6; ++i) {
      const int v  = tid + i * 1024;
      const int bi = v / 192;            // batch row
      const int c4 = v - bi * 192;       // float4 column
      const unsigned long long gp =
          (unsigned long long)(const void*)(xw + ((size_t)(bi * T_ + t)) * G3 + c4 * 4);
      const unsigned lds = xw_lds_base + (unsigned)(bi * G3 + c4 * 4) * 4u;
      asm volatile("global_load_async_to_lds_b128 %0, %1, off"
                   :: "v"(lds), "v"(gp) : "memory");
    }

    // ---- (b) recurrent GEMM: gh = h @ Whh^T ----
#pragma unroll 1
    for (int tile = wave; tile < 96; tile += 32) {   // 3 tiles per wave, uniform
      const int tm = tile / 48;
      const int tn = tile % 48;
      const int arow = tm * 16 + (lane & 15);
      const int g    = tn * 16 + (lane & 15);
      const int kk16 = (lane >> 4) * 16;
      const bf16_t* wrow = Whh + (size_t)g * HD_;
      v8f acc = {};
#pragma unroll
      for (int ks = 0; ks < 8; ++ks) {
        const int k0 = ks * 32;
        v16bf a = make_a_frag(h_s + arow * HD_, k0, lane);   // ds loads
        v16bf b = *(const v16bf*)(wrow + k0 + kk16);         // 32B global load
        acc = wmma_bf16(a, b, acc);
      }
      const int rb  = tm * 16 + (lane >> 4) * 8;
      const int col = tn * 16 + (lane & 15);
#pragma unroll
      for (int i = 0; i < 8; ++i) gh_s[(rb + i) * G3 + col] = acc[i];
    }

    // ---- (c) wait for this wave's async DMA, then workgroup barrier ----
    asm volatile("s_wait_asynccnt 0x0" ::: "memory");
    __syncthreads();

    // ---- gates: r, z, n and state update (xw read from LDS) ----
    for (int e = tid; e < B_ * HD_; e += 1024) {     // 8 elements/thread
      const int bi = e >> 8;
      const int c  = e & 255;
      const float gr = gh_s[bi * G3 + c]        + bhh[c];
      const float gz = gh_s[bi * G3 + 256 + c]  + bhh[256 + c];
      const float gn = gh_s[bi * G3 + 512 + c]  + bhh[512 + c];
      const float r = sigmoidf_(xw_s[bi * G3 + c] + gr);
      const float z = sigmoidf_(xw_s[bi * G3 + 256 + c] + gz);
      const float n = tanhf_(xw_s[bi * G3 + 512 + c] + r * gn);
      const float hp = (float)h_s[e];
      const float hn = (1.f - z) * n + z * hp;
      const bf16_t hb = (bf16_t)hn;
      h_s[e] = hb;
      h_out[((size_t)bi * T_ + t) * HD_ + c] = hb;
    }
    __syncthreads();
  }
}

// ---------------------------------------------------------------------------
// Time-parallel bf16 WMMA GEMM: out[r, col] = A[r,:256] . W[col,:256] + bias[col]
// grid.x * 8 waves cover M-tiles (16 rows each), grid.y covers N-tiles.
// ---------------------------------------------------------------------------
__global__ __launch_bounds__(256) void k_gemm_bf16(
    float* __restrict__ out, const bf16_t* __restrict__ A,
    const bf16_t* __restrict__ W, const float* __restrict__ bias, int ldout) {
  const int lane  = threadIdx.x & 31;
  const int wave  = threadIdx.x >> 5;
  const int mtile = blockIdx.x * 8 + wave;
  const int tn    = blockIdx.y;
  const int arow  = mtile * 16 + (lane & 15);
  const int g     = tn * 16 + (lane & 15);
  const int kk16  = (lane >> 4) * 16;
  const bf16_t* ar = A + (size_t)arow * HD_;
  const bf16_t* wr = W + (size_t)g * HD_;
  v8f acc = {};
#pragma unroll
  for (int ks = 0; ks < 8; ++ks) {
    const int k0 = ks * 32;
    __builtin_prefetch(ar + k0 + 64, 0, 1);   // global_prefetch_b8
    v16bf a = make_a_frag(ar, k0, lane);
    v16bf b = *(const v16bf*)(wr + k0 + kk16);
    acc = wmma_bf16(a, b, acc);
  }
  const int rb  = mtile * 16 + (lane >> 4) * 8;
  const int col = tn * 16 + (lane & 15);
  const float bs = bias[col];
#pragma unroll
  for (int i = 0; i < 8; ++i)
    out[(size_t)(rb + i) * ldout + col] = acc[i] + bs;
}

// ---------------------------------------------------------------------------
// Kalman update: one thread per (b,t). 8x8 Cholesky + per-column solves.
// Only the Joseph-form diagonal is produced (that is all the reference returns).
// ---------------------------------------------------------------------------
__global__ __launch_bounds__(256) void k_kalman(
    float* __restrict__ out, const float* __restrict__ y_seq,
    const float* __restrict__ Hin, const float* __restrict__ Rin,
    const float* __restrict__ head) {
  __shared__ float Hs[M_][N_];
  __shared__ float Rs[M_][M_];
  const int tid = threadIdx.x;
  if (tid < M_ * N_) Hs[tid / N_][tid % N_] = Hin[tid];
  if (tid < M_ * M_) Rs[tid / M_][tid % M_] = Rin[tid];
  __syncthreads();

  const int bt = blockIdx.x * 256 + tid;
  const float* hrow = head + (size_t)bt * 48;

  float mu[N_], p[N_];
#pragma unroll
  for (int n = 0; n < N_; ++n) {
    mu[n] = hrow[n];
    const float lv = hrow[16 + n];
    const float sp = (lv > 20.f) ? lv : log1pf(__expf(lv));
    p[n] = sp + COV_EPS_;
  }
  float y[M_];
#pragma unroll
  for (int m = 0; m < M_; ++m) y[m] = y_seq[(size_t)bt * M_ + m];

  // S = H diag(p) H^T + R (+eps I), lower triangle, then in-place Cholesky
  float L[36];
  {
    int idx = 0;
    for (int i = 0; i < M_; ++i)
      for (int j = 0; j <= i; ++j) {
        float s = Rs[i][j];
        for (int n = 0; n < N_; ++n) s += Hs[i][n] * Hs[j][n] * p[n];
        if (i == j) s += POSDEF_EPS_;
        L[idx++] = s;
      }
  }
  for (int j = 0; j < M_; ++j) {
    float d = L[j * (j + 1) / 2 + j];
    for (int k = 0; k < j; ++k) { const float v = L[j * (j + 1) / 2 + k]; d -= v * v; }
    d = sqrtf(fmaxf(d, 1e-20f));
    L[j * (j + 1) / 2 + j] = d;
    const float inv = 1.f / d;
    for (int i = j + 1; i < M_; ++i) {
      float v = L[i * (i + 1) / 2 + j];
      for (int k = 0; k < j; ++k) v -= L[i * (i + 1) / 2 + k] * L[j * (j + 1) / 2 + k];
      L[i * (i + 1) / 2 + j] = v * inv;
    }
  }

  float innov[M_];
#pragma unroll
  for (int m = 0; m < M_; ++m) {
    float a = y[m];
    for (int n = 0; n < N_; ++n) a -= Hs[m][n] * mu[n];
    innov[m] = a;
  }

  const size_t BTN = (size_t)BT * N_;
  float* o_mu = out;
  float* o_pp = out + BTN;
  float* o_xp = out + 2 * BTN;
  float* o_pd = out + 3 * BTN;
  float* o_yh = out + 4 * BTN;

  float yh[M_];
#pragma unroll
  for (int m = 0; m < M_; ++m) yh[m] = 0.f;

  for (int n = 0; n < N_; ++n) {
    // kn = K[n,:] = (S^{-1} H diag(p))[:,n]
    float kn[M_];
    for (int m = 0; m < M_; ++m) kn[m] = Hs[m][n] * p[n];
    for (int i = 0; i < M_; ++i) {                       // L y = rhs
      float v = kn[i];
      for (int k = 0; k < i; ++k) v -= L[i * (i + 1) / 2 + k] * kn[k];
      kn[i] = v / L[i * (i + 1) / 2 + i];
    }
    for (int i = M_ - 1; i >= 0; --i) {                  // L^T x = y
      float v = kn[i];
      for (int k = i + 1; k < M_; ++k) v -= L[k * (k + 1) / 2 + i] * kn[k];
      kn[i] = v / L[i * (i + 1) / 2 + i];
    }

    float mpn = mu[n];
    for (int m = 0; m < M_; ++m) mpn += kn[m] * innov[m];
    for (int m = 0; m < M_; ++m) yh[m] += Hs[m][n] * mpn;

    o_mu[(size_t)bt * N_ + n] = mu[n];
    o_pp[(size_t)bt * N_ + n] = p[n];
    o_xp[(size_t)bt * N_ + n] = mpn + hrow[32 + n];

    // Joseph diagonal: sum_k (ImKH[n,k])^2 p[k] + kn^T R kn
    float t1 = 0.f;
    for (int k = 0; k < N_; ++k) {
      float c = 0.f;
      for (int m = 0; m < M_; ++m) c += kn[m] * Hs[m][k];
      const float e = ((k == n) ? 1.f : 0.f) - c;
      t1 += e * e * p[k];
    }
    float t2 = 0.f;
    for (int m = 0; m < M_; ++m) {
      float rm = 0.f;
      for (int k = 0; k < M_; ++k) rm += Rs[m][k] * kn[k];
      t2 += kn[m] * rm;
    }
    o_pd[(size_t)bt * N_ + n] = t1 + t2;
  }
#pragma unroll
  for (int m = 0; m < M_; ++m) o_yh[(size_t)bt * M_ + m] = yh[m];
}

// ---------------------------------------------------------------------------
// Launch
// ---------------------------------------------------------------------------
extern "C" void kernel_launch(void* const* d_in, const int* in_sizes, int n_in,
                              void* d_out, int out_size, void* d_ws, size_t ws_size,
                              hipStream_t stream) {
  const float* y_seq = (const float*)d_in[0];
  const float* Hmat  = (const float*)d_in[1];
  const float* Rmat  = (const float*)d_in[2];
  const float* Wih0  = (const float*)d_in[3];
  const float* Whh0  = (const float*)d_in[4];
  const float* bih0  = (const float*)d_in[5];
  const float* bhh0  = (const float*)d_in[6];
  const float* Wih1  = (const float*)d_in[7];
  const float* Whh1  = (const float*)d_in[8];
  const float* bih1  = (const float*)d_in[9];
  const float* bhh1  = (const float*)d_in[10];
  const float* Wm    = (const float*)d_in[11];
  const float* bm    = (const float*)d_in[12];
  const float* Wlv   = (const float*)d_in[13];
  const float* blv   = (const float*)d_in[14];
  const float* Wres  = (const float*)d_in[15];
  const float* bres  = (const float*)d_in[16];
  float* out = (float*)d_out;

  // carve workspace (256B-aligned)
  char* ws = (char*)d_ws;
  size_t off = 0;
  auto carve = [&](size_t bytes) -> void* {
    void* p = ws + off;
    off = (off + bytes + 255) & ~(size_t)255;
    return p;
  };
  bf16_t* whh0_bf  = (bf16_t*)carve((size_t)G3 * HD_ * sizeof(bf16_t));   // 384KB
  bf16_t* wih1_bf  = (bf16_t*)carve((size_t)G3 * HD_ * sizeof(bf16_t));   // 384KB
  bf16_t* whh1_bf  = (bf16_t*)carve((size_t)G3 * HD_ * sizeof(bf16_t));   // 384KB
  bf16_t* whead_bf = (bf16_t*)carve((size_t)48 * HD_ * sizeof(bf16_t));   // 24KB
  float*  bhead    = (float*)carve(48 * sizeof(float));
  bf16_t* h1_bf    = (bf16_t*)carve((size_t)BT * HD_ * sizeof(bf16_t));   // 16MB
  bf16_t* rnn_bf   = (bf16_t*)carve((size_t)BT * HD_ * sizeof(bf16_t));   // 16MB
  float*  xw       = (float*)carve((size_t)BT * G3 * sizeof(float));      // 96MB (reused for both layers)
  float*  headbuf  = (float*)carve((size_t)BT * 48 * sizeof(float));      // 6MB

  const int NW = G3 * HD_;  // 196608
  k_cvt_bf16<<<(NW + 255) / 256, 256, 0, stream>>>(whh0_bf, Whh0, NW);
  k_cvt_bf16<<<(NW + 255) / 256, 256, 0, stream>>>(wih1_bf, Wih1, NW);
  k_cvt_bf16<<<(NW + 255) / 256, 256, 0, stream>>>(whh1_bf, Whh1, NW);
  k_pack_heads<<<(48 * HD_ + 255) / 256, 256, 0, stream>>>(
      whead_bf, bhead, Wm, Wlv, Wres, bm, blv, bres);

  // layer-0 input projection (K=8)
  k_xw0<<<(BT * G3) / 256, 256, 0, stream>>>(xw, y_seq, Wih0, bih0);

  // GRU layer 0 (persistent, WMMA recurrent GEMM + async xw staging)
  k_gru<<<1, 1024, GRU_SMEM, stream>>>(h1_bf, xw, whh0_bf, bhh0);

  // layer-1 input projection: big WMMA GEMM [32768,256]x[256,768]
  k_gemm_bf16<<<dim3(BT / (16 * 8), G3 / 16), 256, 0, stream>>>(
      xw, h1_bf, wih1_bf, bih1, G3);

  // GRU layer 1
  k_gru<<<1, 1024, GRU_SMEM, stream>>>(rnn_bf, xw, whh1_bf, bhh1);

  // heads: [32768,256]x[256,48] WMMA GEMM (Wm|Wlv|Wres)
  k_gemm_bf16<<<dim3(BT / (16 * 8), 48 / 16), 256, 0, stream>>>(
      headbuf, rnn_bf, whead_bf, bhead, 48);

  // Kalman update per (b,t)
  k_kalman<<<BT / 256, 256, 0, stream>>>(out, y_seq, Hmat, Rmat, headbuf);

  (void)in_sizes; (void)n_in; (void)out_size; (void)ws_size;
}
